// Decoder_15994458210943
// MI455X (gfx1250) — matrix operand order repaired and checked by reading, not compiled, not run on hardware
//
#include <hip/hip_runtime.h>
#include <math.h>

// ---------------- problem constants ----------------
#define H_     256
#define O_     63
#define KMIX   10
#define MEM_   20
#define B_     64
#define T_     400
#define L_     256
#define FEAT_  319            // H + O
#define ELEM_  6380           // FEAT * MEM
#define MID_   638
#define KPAD   6400           // ELEM padded to mult of 32
#define KHALF  3200           // K split in two halves across waves
#define NPAD   640            // MID padded to mult of 16
#define NBLK   40             // persistent grid: 40 blocks * 8 waves = 320 waves = 160 tiles x 2 K-halves
#define HIDSZ  (B_ * NPAD)    // one partial hid matrix (40960 floats)
#define COEF_  0.3989422917366028f
#define EPS_   1e-5f
#define ALIGNC 0.05f

typedef __bf16 bf16;
typedef __attribute__((ext_vector_type(8)))  bf16  bf16x8;
typedef __attribute__((ext_vector_type(16))) bf16  bf16x16;
typedef __attribute__((ext_vector_type(8)))  float f32x8;

// ---------------- grid-wide barrier (monotonic counter) ----------------
__device__ __forceinline__ void grid_sync(unsigned* cnt, unsigned target) {
  __threadfence();              // release: push this block's writes to device scope
  __syncthreads();
  if (threadIdx.x == 0) {
    atomicAdd(cnt, 1u);
    while (__hip_atomic_load(cnt, __ATOMIC_RELAXED, __HIP_MEMORY_SCOPE_AGENT) < target)
      __builtin_amdgcn_s_sleep(1);
  }
  __syncthreads();
  __threadfence();              // acquire: invalidate WGP$ so peer writes are visible
}

// ---------------- layer-1 GEMM (partial-K): hid[kh][64x640] = X[:,kh*3200:+3200] @ W^T (+b1 on kh=0)
// bf16 WMMA 16x16x32; 320 waves = 160 tiles x 2 K-halves; ReLU deferred to consumers.
__device__ __forceinline__ void gemm_tile(const bf16* __restrict__ X,
                                          const bf16* __restrict__ W,
                                          const float* __restrict__ b1,
                                          float* __restrict__ hid) {
  const int wave = (blockIdx.x << 3) | (threadIdx.x >> 5);   // 0..319
  const int lane = threadIdx.x & 31;
  const int half = lane >> 4;
  const int r    = lane & 15;
  const int kh   = wave & 1;          // K-half
  const int tile = wave >> 1;         // 0..159
  const int mt   = tile & 3;          // 4 M-tiles (batch)
  const int nt   = tile >> 2;         // 40 N-tiles (neurons)
  const int m0   = mt * 16;
  const int n0   = nt * 16;
  const int k0   = kh * KHALF;

  f32x8 acc = {0.f, 0.f, 0.f, 0.f, 0.f, 0.f, 0.f, 0.f};
  // A layout (16-bit 16x32): lanes 0-15 -> M=r, k {0..7} and {16..23}; lanes 16-31 -> k {8..15},{24..31}
  const bf16* ap = X + (size_t)(m0 + r) * KPAD + k0 + half * 8;
  // B layout (16-bit 32x16): lanes 0-15 -> N=r, k {0..7} and {16..23}; lanes 16-31 -> k {8..15},{24..31}
  const bf16* bp = W + (size_t)(n0 + r) * KPAD + k0 + half * 8;

#pragma unroll 4
  for (int kb = 0; kb < KHALF; kb += 32) {
    bf16x8  alo = *(const bf16x8*)(ap);
    bf16x8  ahi = *(const bf16x8*)(ap + 16);
    bf16x16 a   = __builtin_shufflevector(alo, ahi,
                    0, 1, 2, 3, 4, 5, 6, 7, 8, 9, 10, 11, 12, 13, 14, 15);
    bf16x8  blo = *(const bf16x8*)(bp);
    bf16x8  bhi = *(const bf16x8*)(bp + 16);
    bf16x16 b   = __builtin_shufflevector(blo, bhi,
                    0, 1, 2, 3, 4, 5, 6, 7, 8, 9, 10, 11, 12, 13, 14, 15);
    __builtin_prefetch((const void*)(bp + 128), 0, 1);  // weight stream ahead
    acc = __builtin_amdgcn_wmma_f32_16x16x32_bf16(false, a, false, b,
                                                  (short)0, acc, false, false);
    ap += 32;
    bp += 32;
  }

  const int   n    = n0 + r;
  const float bias = (kh == 0 && n < MID_) ? b1[n] : 0.f;   // bias once (kh=0); ReLU in consumer
  float* hp = hid + (size_t)kh * HIDSZ;
#pragma unroll
  for (int j = 0; j < 8; ++j) {     // C layout: lane -> N, VGPR j -> M = j + 8*half
    hp[(size_t)(m0 + j + half * 8) * NPAD + n] = acc[j] + bias;
  }
}

// combine the two K-partials + ReLU
__device__ __forceinline__ float hid_relu(const float* __restrict__ hid, int k) {
  float v = hid[k] + hid[k + HIDSZ];
  return v > 0.f ? v : 0.f;
}

// ---------------- phase 2: attention MLP head + Graves mixture + buffer shift ----------------
__device__ void phase2(int t, const float* __restrict__ x, const float* __restrict__ context,
                       const float* __restrict__ na_w2, const float* __restrict__ na_b2,
                       const float* __restrict__ hid_a, float* __restrict__ mu,
                       const float* __restrict__ idt,
                       const bf16* __restrict__ Xc, bf16* __restrict__ Xn,
                       float* __restrict__ attns, float* smem) {
  if (blockIdx.x >= 8) return;                       // waves 0..63 <-> batch rows
  const int widx = threadIdx.x >> 5;
  const int lane = threadIdx.x & 31;
  const int b    = (blockIdx.x << 3) | widx;
  const float* ha = hid_a + (size_t)b * NPAD;

  float gbk[30];
#pragma unroll
  for (int j = 0; j < 30; ++j) {
    const float* w = na_w2 + (size_t)j * MID_;
    float p = 0.f;
    for (int k = lane; k < MID_; k += 32) p += hid_relu(ha, k) * w[k];
#pragma unroll
    for (int off = 16; off; off >>= 1) p += __shfl_xor(p, off, 32);
    gbk[j] = p + na_b2[j];
  }

  // softmax(gbk[0:10]) + EPS ; sig = exp(gbk[10:20]) + EPS ; mu += ALIGN*exp(gbk[20:30])
  float mx = gbk[0];
#pragma unroll
  for (int k = 1; k < KMIX; ++k) mx = fmaxf(mx, gbk[k]);
  float es[KMIX], sum = 0.f;
#pragma unroll
  for (int k = 0; k < KMIX; ++k) { es[k] = expf(gbk[k] - mx); sum += es[k]; }
  float g[KMIX], sig[KMIX], munew[KMIX];
  const float inv = 1.f / sum;
#pragma unroll
  for (int k = 0; k < KMIX; ++k) {
    g[k]     = es[k] * inv + EPS_;
    sig[k]   = expf(gbk[KMIX + k]) + EPS_;
    munew[k] = mu[b * KMIX + k] + ALIGNC * expf(gbk[2 * KMIX + k]);
  }
  if (lane == 0) {
#pragma unroll
    for (int k = 0; k < KMIX; ++k) mu[b * KMIX + k] = munew[k];
  }

  // alpha[l] = COEF * sum_k g*exp(-0.5*sig*(mu-l)^2)
  float* s_alpha = smem + widx * 256;
#pragma unroll
  for (int i = 0; i < 8; ++i) {
    const int l = lane + 32 * i;
    float s = 0.f;
#pragma unroll
    for (int k = 0; k < KMIX; ++k) {
      float d = munew[k] - (float)l;
      s += g[k] * expf(-0.5f * sig[k] * d * d);
    }
    float al = COEF_ * s;
    s_alpha[l] = al;
    attns[((size_t)t * B_ + b) * L_ + l] = al;
  }
  __syncthreads();

  // c_t[h] = sum_l alpha[l] * ctx[b,l,h]   (context is [L,B,H])
  float c[8] = {0.f, 0.f, 0.f, 0.f, 0.f, 0.f, 0.f, 0.f};
  for (int l = 0; l < L_; ++l) {
    const float a   = s_alpha[l];
    const float* cp = context + ((size_t)l * B_ + b) * H_;
#pragma unroll
    for (int i = 0; i < 8; ++i) c[i] += a * cp[lane + 32 * i];
  }

  // z column (m=0): first H rows = c_t + idt, next O rows = x[t]/30
  bf16* xr = Xn + (size_t)b * KPAD;
#pragma unroll
  for (int i = 0; i < 8; ++i) {
    const int f = lane + 32 * i;
    xr[f * MEM_] = (bf16)(c[i] + idt[b * H_ + f]);
  }
  for (int i = lane; i < O_; i += 32)
    xr[(H_ + i) * MEM_] = (bf16)(x[((size_t)t * B_ + b) * O_ + i] * (1.f / 30.f));

  // shift register: Xn[f*20+m] = Xc[f*20+m-1], m>=1
  const bf16* xc = Xc + (size_t)b * KPAD;
  for (int e = lane; e < ELEM_; e += 32) {
    if (e % MEM_) xr[e] = xc[e - 1];
  }
}

// ---------------- phase 4: u = nu layer-2 (+idt on first H), overwrite column 0 ----------------
__device__ void phase4(const float* __restrict__ hid_u, const float* __restrict__ nu_w2,
                       const float* __restrict__ nu_b2, const float* __restrict__ idt,
                       bf16* __restrict__ Xn) {
  const int gt = blockIdx.x * 256 + threadIdx.x;         // 10240 threads, 64*319 outputs
  for (int idx = gt; idx < B_ * FEAT_; idx += NBLK * 256) {
    const int b = idx / FEAT_;
    const int f = idx % FEAT_;
    const float* hu = hid_u + (size_t)b * NPAD;
    const float* w  = nu_w2 + (size_t)f * MID_;
    float s = nu_b2[f];
    for (int k = 0; k < MID_; ++k) s += hid_relu(hu, k) * w[k];
    if (f < H_) s += idt[b * H_ + f];
    Xn[(size_t)b * KPAD + f * MEM_] = (bf16)s;
  }
}

// ---------------- phase 6: output head ----------------
__device__ void phase6(int t, const float* __restrict__ hid_o, const float* __restrict__ no_w2,
                       const float* __restrict__ no_b2, const float* __restrict__ sp_out,
                       const float* __restrict__ out_w, const float* __restrict__ out_b,
                       float* __restrict__ out_seq, float* smem) {
  if (blockIdx.x >= 8) return;
  const int widx = threadIdx.x >> 5;
  const int lane = threadIdx.x & 31;
  const int b    = (blockIdx.x << 3) | widx;
  const float* ho = hid_o + (size_t)b * NPAD;
  float* s_ot = smem + widx * 256;
#pragma unroll
  for (int i = 0; i < 8; ++i) {
    const int h = lane + 32 * i;
    const float* w = no_w2 + (size_t)h * MID_;
    float s = no_b2[h];
    for (int k = 0; k < MID_; ++k) s += hid_relu(ho, k) * w[k];
    s_ot[h] = s + sp_out[b * H_ + h];
  }
  __syncthreads();
  for (int o = lane; o < O_; o += 32) {
    const float* w = out_w + (size_t)o * H_;
    float s = out_b[o];
    for (int h = 0; h < H_; ++h) s += s_ot[h] * w[h];
    out_seq[((size_t)t * B_ + b) * O_ + o] = s;
  }
}

// ---------------- init kernels ----------------
__global__ void k_convert_weights(const float* __restrict__ wa, const float* __restrict__ wu,
                                  const float* __restrict__ wo,
                                  bf16* __restrict__ Wa, bf16* __restrict__ Wu,
                                  bf16* __restrict__ Wo) {
  const long per   = (long)NPAD * KPAD;
  const long total = 3L * per;
  for (long i = (long)blockIdx.x * blockDim.x + threadIdx.x; i < total;
       i += (long)gridDim.x * blockDim.x) {
    const int  which = (int)(i / per);
    const long rem   = i % per;
    const int  n = (int)(rem / KPAD);
    const int  k = (int)(rem % KPAD);
    const float* src = which == 0 ? wa : which == 1 ? wu : wo;
    bf16*        dst = which == 0 ? Wa : which == 1 ? Wu : Wo;
    float v = (n < MID_ && k < ELEM_) ? src[(long)n * ELEM_ + k] : 0.f;
    dst[rem] = (bf16)v;
  }
}

__global__ void k_init_state(const float* __restrict__ ident,
                             const float* __restrict__ fu_w, const float* __restrict__ fu_b,
                             const float* __restrict__ fo_w, const float* __restrict__ fo_b,
                             bf16* __restrict__ X0, bf16* __restrict__ X1,
                             float* __restrict__ mu, unsigned* __restrict__ cnt,
                             float* __restrict__ idt, float* __restrict__ sp_out) {
  const int b = blockIdx.x;        // 64 blocks
  const int h = threadIdx.x;       // 256 threads
  // idt / sp_out
  {
    float s1 = fu_b[h], s2 = fo_b[h];
    const float* ir = ident + (size_t)b * H_;
    for (int j = 0; j < H_; ++j) {
      s1 += ir[j] * fu_w[(size_t)h * H_ + j];
      s2 += ir[j] * fo_w[(size_t)h * H_ + j];
    }
    idt[b * H_ + h]    = tanhf(s1);
    sp_out[b * H_ + h] = s2;
  }
  // S0: first H feature rows hold ident across all MEM columns; rest zero; pad zero
  for (int e = h; e < KPAD; e += 256) {
    float v = 0.f;
    if (e < ELEM_) {
      const int f = e / MEM_;
      if (f < H_) v = ident[(size_t)b * H_ + f];
    }
    X0[(size_t)b * KPAD + e] = (bf16)v;
    X1[(size_t)b * KPAD + e] = (bf16)0.f;
  }
  if (b == 0) {
    for (int i = h; i < B_ * KMIX; i += 256) mu[i] = 0.f;
    if (h == 0) *cnt = 0u;
  }
}

// ---------------- persistent decoder kernel ----------------
__global__ __launch_bounds__(256) void k_decoder(
    const float* __restrict__ x, const float* __restrict__ context,
    const float* __restrict__ na_b1, const float* __restrict__ na_w2, const float* __restrict__ na_b2,
    const float* __restrict__ nu_b1, const float* __restrict__ nu_w2, const float* __restrict__ nu_b2,
    const float* __restrict__ no_b1, const float* __restrict__ no_w2, const float* __restrict__ no_b2,
    const float* __restrict__ out_w, const float* __restrict__ out_b,
    const bf16* __restrict__ Wa, const bf16* __restrict__ Wu, const bf16* __restrict__ Wo,
    bf16* X0, bf16* X1,
    float* hid_a, float* hid_u, float* hid_o,
    const float* __restrict__ idt, const float* __restrict__ sp_out,
    float* mu, unsigned* cnt, float* out_seq, float* attns) {
  __shared__ float smem[8 * 256];
  bf16* Xc = X0;
  bf16* Xn = X1;
  unsigned bars = 0;

  for (int t = 0; t < T_; ++t) {
    // P1: attention MLP layer-1 on S
    gemm_tile(Xc, Wa, na_b1, hid_a);
    grid_sync(cnt, ++bars * NBLK);
    // P2: mixture attention, build Sp = [z, S[:, :-1]]
    phase2(t, x, context, na_w2, na_b2, hid_a, mu, idt, Xc, Xn, attns, smem);
    grid_sync(cnt, ++bars * NBLK);
    // P3: update MLP layer-1 on Sp
    gemm_tile(Xn, Wu, nu_b1, hid_u);
    grid_sync(cnt, ++bars * NBLK);
    // P4: u (+idt), S_new column 0
    phase4(hid_u, nu_w2, nu_b2, idt, Xn);
    grid_sync(cnt, ++bars * NBLK);
    // P5: output MLP layer-1 on S_new
    gemm_tile(Xn, Wo, no_b1, hid_o);
    grid_sync(cnt, ++bars * NBLK);
    // P6: output projection
    phase6(t, hid_o, no_w2, no_b2, sp_out, out_w, out_b, out_seq, smem);
    grid_sync(cnt, ++bars * NBLK);
    // swap state buffers: S <- S_new
    bf16* tmp = Xc; Xc = Xn; Xn = tmp;
  }
}

// ---------------- host ----------------
extern "C" void kernel_launch(void* const* d_in, const int* in_sizes, int n_in,
                              void* d_out, int out_size, void* d_ws, size_t ws_size,
                              hipStream_t stream) {
  const float* x       = (const float*)d_in[0];
  const float* ident   = (const float*)d_in[1];
  const float* context = (const float*)d_in[2];
  const float* na_w1   = (const float*)d_in[3];
  const float* na_b1   = (const float*)d_in[4];
  const float* na_w2   = (const float*)d_in[5];
  const float* na_b2   = (const float*)d_in[6];
  const float* no_w1   = (const float*)d_in[7];
  const float* no_b1   = (const float*)d_in[8];
  const float* no_w2   = (const float*)d_in[9];
  const float* no_b2   = (const float*)d_in[10];
  const float* nu_w1   = (const float*)d_in[11];
  const float* nu_b1   = (const float*)d_in[12];
  const float* nu_w2   = (const float*)d_in[13];
  const float* nu_b2   = (const float*)d_in[14];
  const float* out_w   = (const float*)d_in[15];
  const float* out_b   = (const float*)d_in[16];
  const float* fu_w    = (const float*)d_in[17];
  const float* fu_b    = (const float*)d_in[18];
  const float* fo_w    = (const float*)d_in[19];
  const float* fo_b    = (const float*)d_in[20];

  char* ws = (char*)d_ws;
  const size_t szW   = (size_t)NPAD * KPAD * sizeof(bf16);     // 8,192,000
  const size_t szX   = (size_t)B_ * KPAD * sizeof(bf16);       //   819,200
  const size_t szHid = (size_t)2 * HIDSZ * sizeof(float);      //   327,680 (2 K-partials)
  const size_t szIdt = (size_t)B_ * H_ * sizeof(float);        //    65,536
  size_t off = 0;
  bf16*  Wa     = (bf16*)(ws + off); off += szW;
  bf16*  Wu     = (bf16*)(ws + off); off += szW;
  bf16*  Wo     = (bf16*)(ws + off); off += szW;
  bf16*  X0     = (bf16*)(ws + off); off += szX;
  bf16*  X1     = (bf16*)(ws + off); off += szX;
  float* hid_a  = (float*)(ws + off); off += szHid;
  float* hid_u  = (float*)(ws + off); off += szHid;
  float* hid_o  = (float*)(ws + off); off += szHid;
  float* idt    = (float*)(ws + off); off += szIdt;
  float* sp_out = (float*)(ws + off); off += szIdt;
  float* mu     = (float*)(ws + off); off += 4096;             // B*KMIX floats (2560 B), padded
  unsigned* cnt = (unsigned*)(ws + off); off += 256;

  float* out_seq = (float*)d_out;
  float* attns   = out_seq + (size_t)T_ * B_ * O_;

  k_convert_weights<<<2048, 256, 0, stream>>>(na_w1, nu_w1, no_w1, Wa, Wu, Wo);
  k_init_state<<<B_, 256, 0, stream>>>(ident, fu_w, fu_b, fo_w, fo_b,
                                       X0, X1, mu, cnt, idt, sp_out);
  k_decoder<<<NBLK, 256, 0, stream>>>(x, context,
                                      na_b1, na_w2, na_b2,
                                      nu_b1, nu_w2, nu_b2,
                                      no_b1, no_w2, no_b2,
                                      out_w, out_b,
                                      Wa, Wu, Wo, X0, X1,
                                      hid_a, hid_u, hid_o,
                                      idt, sp_out, mu, cnt, out_seq, attns);
}
